// MultiHeadedGraphAttention_72138270703637
// MI455X (gfx1250) — matrix operand order repaired and checked by reading, not compiled, not run on hardware
//
#include <hip/hip_runtime.h>
#include <hip/hip_bf16.h>
#include <cstdint>

typedef __attribute__((ext_vector_type(16))) __bf16 v16bf;
typedef __attribute__((ext_vector_type(8)))  __bf16 v8bf;
typedef __attribute__((ext_vector_type(8)))  float  v8f;
typedef __attribute__((ext_vector_type(4)))  float  v4f;
typedef __attribute__((ext_vector_type(4)))  unsigned int v4u;
typedef __attribute__((ext_vector_type(8)))  int v8i;
typedef __attribute__((ext_vector_type(4)))  int v4i;

#define BB  8
#define MM  1024
#define DD  1024
#define HH  16
#define HDD 64

#if defined(__HIP_DEVICE_COMPILE__) && __has_builtin(__builtin_amdgcn_tensor_load_to_lds) && \
    __has_builtin(__builtin_amdgcn_s_wait_tensorcnt)
#define HAVE_TDM 1
#else
#define HAVE_TDM 0
#endif

// ---------------------------------------------------------------------------
// WMMA helpers (CDNA5 16x16x32 bf16, f32 accumulate).
// Fragment layouts as loaded by ld_frag_a/ld_frag_b (A and B share one k order):
//   A (16x32 bf16): lane<16 -> row=lane,    ks 0..15 (contiguous)
//                   lane>=16 -> row=lane-16, ks 16..31
//   B (32x16 bf16): lane<16 -> col=lane,    ks 0..15 (contiguous)
//                   lane>=16 -> col=lane-16, ks 16..31
//   C/D (16x16 f32): m = vgpr + 8*(lane>=16), n = lane&15
// ---------------------------------------------------------------------------
__device__ __forceinline__ v8f wmma_bf16(v16bf a, v16bf b, v8f c) {
  return __builtin_amdgcn_wmma_f32_16x16x32_bf16(
      /*neg_a=*/false, a, /*neg_b=*/false, b,
      /*c_mod=*/(short)0, c, /*reuse_a=*/false, /*reuse_b=*/false);
}

// rowptr points at 32 contiguous bf16 (one A row, one 32-wide k-chunk)
__device__ __forceinline__ v16bf ld_frag_a(const __bf16* rowptr, int hi) {
  const int o = hi ? 16 : 0;
  v8bf lo = *(const v8bf*)(rowptr + o);
  v8bf hh = *(const v8bf*)(rowptr + o + 8);
  v16bf r;
#pragma unroll
  for (int i = 0; i < 8; ++i) { r[i] = lo[i]; r[i + 8] = hh[i]; }
  return r;
}

// colptr points at 32 contiguous bf16 holding B[k=0..31, n] for this lane's n
__device__ __forceinline__ v16bf ld_frag_b(const __bf16* colptr, int hi) {
  const int o = hi ? 16 : 0;
  v8bf lo = *(const v8bf*)(colptr + o);
  v8bf hh = *(const v8bf*)(colptr + o + 8);
  v16bf r;
#pragma unroll
  for (int i = 0; i < 8; ++i) { r[i] = lo[i]; r[i + 8] = hh[i]; }
  return r;
}

// ---------------------------------------------------------------------------
// Tensor Data Mover: 2D bf16 tile load, descriptor per CDNA5 ISA.
//   group0: count=1 | lds_addr | global_addr[56:0] | type=2
//   group1: data_size=1(2B) | tensor_dim0/1 | tile_dim0/1 | dim0 stride
//   groups 2/3 zero (2D tensor)
// ---------------------------------------------------------------------------
#if HAVE_TDM
__device__ __forceinline__ void tdm_load_tile_2d(const void* gsrc, unsigned lds_off,
                                                 unsigned w, unsigned h, unsigned stride) {
  const unsigned long long ga = (unsigned long long)(uintptr_t)gsrc;
  v4u g0;
  g0[0] = 1u;                                                   // count=1, user D#
  g0[1] = lds_off;                                              // lds_addr (bytes)
  g0[2] = (unsigned)ga;                                         // global_addr[31:0]
  g0[3] = (unsigned)((ga >> 32) & 0x01ffffffu) | (2u << 30);    // ga[56:32] | type=2
  v8i g1;
  g1[0] = (int)(1u << 16);                                      // data_size = 2 bytes
  g1[1] = (int)((w & 0xffffu) << 16);                           // tensor_dim0 lo16
  g1[2] = (int)((w >> 16) | ((h & 0xffffu) << 16));             // dim0 hi | dim1 lo
  g1[3] = (int)(((h >> 16) & 0xffffu) | ((w & 0xffffu) << 16)); // dim1 hi | tile_dim0
  g1[4] = (int)(h & 0xffffu);                                   // tile_dim1 (tile_dim2=0)
  g1[5] = (int)stride;                                          // tensor_dim0_stride lo
  g1[6] = 0;
  g1[7] = 0;
  v4i g2 = {0, 0, 0, 0};
  v4i g3 = {0, 0, 0, 0};
#if defined(__clang_major__) && (__clang_major__ >= 23)
  v8i g4 = {0, 0, 0, 0, 0, 0, 0, 0};
  __builtin_amdgcn_tensor_load_to_lds(g0, g1, g2, g3, g4, 0);
#else
  __builtin_amdgcn_tensor_load_to_lds(g0, g1, g2, g3, 0);
#endif
}
#endif

// ---------------------------------------------------------------------------
// pos_sum = shared_rl_pe.sum(0) : [M,M,HD] -> [M,HD].  Pure bandwidth (256MB).
// ---------------------------------------------------------------------------
__global__ void zero_kernel(float* p, int n) {
  int i = blockIdx.x * blockDim.x + threadIdx.x;
  if (i < n) p[i] = 0.f;
}

__global__ __launch_bounds__(256) void possum_kernel(
    const float* __restrict__ rl, float* __restrict__ pos) {
  const int out = blockIdx.x * 256 + threadIdx.x;  // 0..65535 (gridDim.x = 256)
  const int i0  = blockIdx.y * 128;                // gridDim.y = 8
  float s = 0.f;
#pragma unroll 4
  for (int i = 0; i < 128; ++i)
    s += rl[(size_t)(i0 + i) * (MM * HDD) + out];
  atomicAdd(&pos[out], s);
}

// ---------------------------------------------------------------------------
// Tiled WMMA GEMM: C[8192,1024] = A_f32 @ W_f32 (+bias)(+pos_sum)(*scale)
// BM=BN=BK=64. 8 waves, each owns a 16x32 strip -> 4 wmma per barrier pair.
// mode 0: f32 flat [r*1024+c] (final Wo GEMM)
// mode 1: bf16 out in [B,H,M,HD] layout (projections, ready for attention)
// ---------------------------------------------------------------------------
__global__ __launch_bounds__(256) void gemm_proj_kernel(
    const float* __restrict__ A, const float* __restrict__ W,
    const float* __restrict__ bias, const float* __restrict__ pos_sum,
    float scale, int mode, float* __restrict__ outF, __bf16* __restrict__ outB)
{
  __shared__ __align__(16) __bf16 As[64][72];  // [m][k], padded rows (144B)
  __shared__ __align__(16) __bf16 Bs[64][72];  // [n][k] (W tile transposed)

  const int K = DD, N = HH * HDD;
  const int mBase = blockIdx.x * 64, nBase = blockIdx.y * 64;
  const int tid = threadIdx.x, lane = tid & 31, wave = tid >> 5;
  const int hi = lane >> 4, ln = lane & 15;
  const int wm = (wave & 3) * 16, wn = (wave >> 2) * 32;

  v8f acc0 = {}; v8f acc1 = {};

  const int ra = tid >> 2, ka = (tid & 3) * 16;  // A staging: 4 thr/row, 16 f32 each
  const int nb = (tid & 3) * 16;                 // W staging: 4 thr/row, 16 f32 each
  const float* aPtr = A + (size_t)(mBase + ra) * K + ka;
  const float* wPtr = W + (size_t)ra * N + nBase + nb;

  for (int k0 = 0; k0 < K; k0 += 64) {
#pragma unroll
    for (int j = 0; j < 16; j += 4) {
      v4f av = *(const v4f*)(aPtr + k0 + j);
      As[ra][ka + j + 0] = (__bf16)av[0];
      As[ra][ka + j + 1] = (__bf16)av[1];
      As[ra][ka + j + 2] = (__bf16)av[2];
      As[ra][ka + j + 3] = (__bf16)av[3];
      v4f wv = *(const v4f*)(wPtr + (size_t)k0 * N + j);
      Bs[nb + j + 0][ra] = (__bf16)wv[0];
      Bs[nb + j + 1][ra] = (__bf16)wv[1];
      Bs[nb + j + 2][ra] = (__bf16)wv[2];
      Bs[nb + j + 3][ra] = (__bf16)wv[3];
    }
    __builtin_prefetch(aPtr + k0 + 64, 0, 1);                 // global_prefetch_b8
    __builtin_prefetch(wPtr + (size_t)(k0 + 64) * N, 0, 1);
    __syncthreads();

#pragma unroll
    for (int kc = 0; kc < 2; ++kc) {
      v16bf a  = ld_frag_a(&As[wm + ln][kc * 32], hi);
      v16bf b0 = ld_frag_b(&Bs[wn + ln][kc * 32], hi);
      v16bf b1 = ld_frag_b(&Bs[wn + 16 + ln][kc * 32], hi);
      acc0 = wmma_bf16(a, b0, acc0);
      acc1 = wmma_bf16(a, b1, acc1);
    }
    __syncthreads();
  }

#pragma unroll
  for (int nt = 0; nt < 2; ++nt) {
    v8f acc = nt ? acc1 : acc0;
#pragma unroll
    for (int i = 0; i < 8; ++i) {
      const int r = mBase + wm + i + 8 * hi;
      const int c = nBase + wn + nt * 16 + ln;
      const int m = r & (MM - 1);
      float val = acc[i] + bias[c];
      if (pos_sum) val += pos_sum[m * HDD + (c & (HDD - 1))];
      val *= scale;
      if (mode == 0) {
        outF[(size_t)r * N + c] = val;
      } else {
        const int b = r >> 10, h = c >> 6, d = c & 63;
        outB[((size_t)(b * HH + h) * MM + m) * HDD + d] = (__bf16)val;
      }
    }
  }
}

// ---------------------------------------------------------------------------
// Attention per (b, h, 32-row slab).  pos_sum is pre-folded into K',V'.
//   Phase 1: S[32,1024] = Q' K'^T  (WMMA; Q frags register-resident)
//   Phase 2: masked softmax (8 threads/row, __shfl_xor), P -> bf16 in LDS
//   Phase 3: ctx[32,64] = P @ V'.  V tiles arrive via double-buffered TDM
//            (tensor_load_to_lds + s_wait_tensorcnt) so the DMA engine fills
//            LDS while all 8 waves keep issuing WMMAs.
// Dynamic LDS = 128KB (S) + 64KB (P) + 16KB (2x V tile) < 320KB/WGP.
// ---------------------------------------------------------------------------
#define SF_BYTES (32 * 1024 * 4)
#define PA_BYTES (32 * 1024 * 2)
#define VD_OFF   (SF_BYTES + PA_BYTES)
#define ATTN_SMEM (VD_OFF + 2 * 64 * 64 * 2)

__global__ __launch_bounds__(256) void attn_kernel(
    const __bf16* __restrict__ Qb, const __bf16* __restrict__ Kb,
    const __bf16* __restrict__ Vb, const unsigned char* __restrict__ mask,
    float* __restrict__ ctx)
{
  extern __shared__ __align__(16) char smem[];
  float*  Sf = (float*)smem;                    // [32][1024]
  __bf16* Pa = (__bf16*)(smem + SF_BYTES);      // [32][1024]

  const int slab = blockIdx.x;          // 0..31
  const int bh   = blockIdx.y;          // 0..127
  const int b = bh >> 4, h = bh & 15;
  const int r0 = slab * 32;
  const size_t base = (size_t)bh * MM * HDD;
  const __bf16* Q = Qb + base;
  const __bf16* K = Kb + base;
  const __bf16* V = Vb + base;

  const int tid = threadIdx.x, lane = tid & 31, wave = tid >> 5;
  const int hi = lane >> 4, ln = lane & 15;

  // ---- Phase 1: scores -------------------------------------------------
  {
    const int rt = wave & 1;              // 16-row tile within slab
    const int cbase = (wave >> 1) * 256;  // 256-column strip per wave pair
    const int qrow = r0 + rt * 16 + ln;
    v16bf aq0 = ld_frag_a(Q + (size_t)qrow * HDD + 0,  hi);  // HD ks 0..31
    v16bf aq1 = ld_frag_a(Q + (size_t)qrow * HDD + 32, hi);  // HD ks 32..63
    for (int ct = 0; ct < 16; ++ct) {
      const int c0 = cbase + ct * 16;
      __builtin_prefetch(K + (size_t)(c0 + 16 + ln) * HDD, 0, 1);
      v16bf bk0 = ld_frag_b(K + (size_t)(c0 + ln) * HDD + 0,  hi);
      v16bf bk1 = ld_frag_b(K + (size_t)(c0 + ln) * HDD + 32, hi);
      v8f s = {};
      s = wmma_bf16(aq0, bk0, s);
      s = wmma_bf16(aq1, bk1, s);
#pragma unroll
      for (int i = 0; i < 8; ++i)
        Sf[(rt * 16 + i + 8 * hi) * 1024 + c0 + ln] = s[i];
    }
  }
  __syncthreads();

  // ---- Phase 2: masked softmax ----------------------------------------
  {
    const int row = tid >> 3;
    const int seg = (tid & 7) * 128;
    float* srow = Sf + row * 1024 + seg;
    const unsigned char* mrow = mask + b * MM + seg;
    float mx = -1e30f;
    for (int j = 0; j < 128; ++j)
      mx = fmaxf(mx, mrow[j] ? srow[j] : -1e30f);
    mx = fmaxf(mx, __shfl_xor(mx, 1));
    mx = fmaxf(mx, __shfl_xor(mx, 2));
    mx = fmaxf(mx, __shfl_xor(mx, 4));
    float sum = 0.f;
    for (int j = 0; j < 128; ++j) {
      float e = mrow[j] ? __expf(srow[j] - mx) : 0.f;
      srow[j] = e;
      sum += e;
    }
    sum += __shfl_xor(sum, 1);
    sum += __shfl_xor(sum, 2);
    sum += __shfl_xor(sum, 4);
    const float inv = 1.f / sum;
    __bf16* prow = Pa + row * 1024 + seg;
    for (int j = 0; j < 128; ++j) prow[j] = (__bf16)(srow[j] * inv);
  }
  __syncthreads();

  // ---- Phase 3: context = P @ V' --------------------------------------
  const int rt = wave & 1;    // output row tile
  const int ct = wave >> 1;   // output col tile (0..3 over HD=64)
  v8f acc = {};

#if HAVE_TDM
  {
    __bf16* Vd = (__bf16*)(smem + VD_OFF);   // 2 x [64][64] bf16 (row-major k,n)
    if (wave == 0)  // pre-issue chunk 0 (TDM ignores EXEC; one issue per wave)
      tdm_load_tile_2d(V, VD_OFF, 64, 64, 64);
    for (int c = 0; c < 16; ++c) {
      if (wave == 0) {
        if (c + 1 < 16) {
          tdm_load_tile_2d(V + (size_t)(c + 1) * 64 * 64,
                           VD_OFF + ((c + 1) & 1) * 8192, 64, 64, 64);
          __builtin_amdgcn_s_wait_tensorcnt(1);  // in-order: chunk c resident
        } else {
          __builtin_amdgcn_s_wait_tensorcnt(0);
        }
      }
      __syncthreads();
      const __bf16* vb = Vd + (c & 1) * 4096;
#pragma unroll
      for (int kc = 0; kc < 2; ++kc) {
        const int k0 = c * 64 + kc * 32;
        v16bf a = ld_frag_a(Pa + (rt * 16 + ln) * 1024 + k0, hi);
        // transpose-on-read: B[k][n] = vb[kc*32+k][ct*16+n]
        const __bf16* col = vb + (kc * 32 + (hi ? 16 : 0)) * 64 + ct * 16 + ln;
        v16bf bv;
#pragma unroll
        for (int i = 0; i < 16; ++i) bv[i] = col[i * 64];
        acc = wmma_bf16(a, bv, acc);
      }
      __syncthreads();
    }
  }
#else
  {
    __bf16* Vt = (__bf16*)(smem + VD_OFF);   // [64][40] transposed stage
    const int kk = tid & 31, nn = (tid >> 5) * 8;
    for (int k0 = 0; k0 < MM; k0 += 32) {
      const __bf16* vsrc = V + (size_t)(k0 + kk) * HDD + nn;
#pragma unroll
      for (int j = 0; j < 8; ++j) Vt[(nn + j) * 40 + kk] = vsrc[j];
      __syncthreads();
      v16bf a  = ld_frag_a(Pa + (rt * 16 + ln) * 1024 + k0, hi);
      v16bf bv = ld_frag_b(&Vt[(ct * 16 + ln) * 40], hi);
      acc = wmma_bf16(a, bv, acc);
      __syncthreads();
    }
  }
#endif

#pragma unroll
  for (int i = 0; i < 8; ++i) {
    const int m = r0 + rt * 16 + i + 8 * hi;
    const int c = h * HDD + ct * 16 + ln;
    ctx[((size_t)b * MM + m) * (HH * HDD) + c] = acc[i];
  }
}

// ---------------------------------------------------------------------------
extern "C" void kernel_launch(void* const* d_in, const int* in_sizes, int n_in,
                              void* d_out, int out_size, void* d_ws, size_t ws_size,
                              hipStream_t stream) {
  (void)in_sizes; (void)n_in; (void)out_size; (void)ws_size;
  const float* k  = (const float*)d_in[0];
  const float* v  = (const float*)d_in[1];
  const float* q  = (const float*)d_in[2];
  const unsigned char* mask = (const unsigned char*)d_in[3];
  const float* rl = (const float*)d_in[4];
  const float* Wk = (const float*)d_in[5];
  const float* bk = (const float*)d_in[6];
  const float* Wv = (const float*)d_in[7];
  const float* bv = (const float*)d_in[8];
  const float* Wq = (const float*)d_in[9];
  const float* bq = (const float*)d_in[10];
  const float* Wo = (const float*)d_in[11];
  const float* bo = (const float*)d_in[12];
  float* out = (float*)d_out;

  // workspace: pos_sum(f32) | Q'(bf16) | K'(bf16) | V'(bf16) | ctx(f32)
  char* ws = (char*)d_ws;
  size_t off = 0;
  float*  pos = (float*)(ws + off);  off += (size_t)MM * HDD * sizeof(float);
  __bf16* Qb  = (__bf16*)(ws + off); off += (size_t)BB * HH * MM * HDD * 2;
  __bf16* Kb  = (__bf16*)(ws + off); off += (size_t)BB * HH * MM * HDD * 2;
  __bf16* Vb  = (__bf16*)(ws + off); off += (size_t)BB * HH * MM * HDD * 2;
  float*  ctx = (float*)(ws + off);

  zero_kernel<<<dim3((MM * HDD + 255) / 256), 256, 0, stream>>>(pos, MM * HDD);
  possum_kernel<<<dim3(256, 8), 256, 0, stream>>>(rl, pos);

  const dim3 gg(BB * MM / 64, (HH * HDD) / 64);
  // K' = k@Wk + bk + pos_sum ; V' = v@Wv + bv + pos_sum ; Q' = (q@Wq+bq)/8
  gemm_proj_kernel<<<gg, 256, 0, stream>>>(k, Wk, bk, pos, 1.0f,   1, nullptr, Kb);
  gemm_proj_kernel<<<gg, 256, 0, stream>>>(v, Wv, bv, pos, 1.0f,   1, nullptr, Vb);
  gemm_proj_kernel<<<gg, 256, 0, stream>>>(q, Wq, bq, nullptr, 0.125f, 1, nullptr, Qb);

  hipFuncSetAttribute((const void*)attn_kernel,
                      hipFuncAttributeMaxDynamicSharedMemorySize, ATTN_SMEM);
  attn_kernel<<<dim3(MM / 32, BB * HH), 256, ATTN_SMEM, stream>>>(Qb, Kb, Vb, mask, ctx);

  gemm_proj_kernel<<<gg, 256, 0, stream>>>(ctx, Wo, bo, nullptr, 1.0f, 0, out, nullptr);
}
